// ScoreNet2D_25417616458104
// MI455X (gfx1250) — matrix operand order repaired and checked by reading, not compiled, not run on hardware
//
#include <hip/hip_runtime.h>
#include <hip/hip_bf16.h>
#include <math.h>

typedef __attribute__((ext_vector_type(16))) _Float16 v16h;
typedef __attribute__((ext_vector_type(8)))  _Float16 v8h;
typedef __attribute__((ext_vector_type(8)))  float    v8f;

union AV { v16h v; _Float16 h[16]; };
union CV { v8f  v; float f[8]; };

static constexpr int   Bn = 64, Hn = 128, Wn = 128, En = 256;   // E = 2W
static constexpr float TWO_PI_F = 6.28318530717958647692f;
static constexpr float GN_EPS = 1e-5f;

static inline int cdiv(long long a, long long b) { return (int)((a + b - 1) / b); }

// ---------------------------------------------------------------------------
// 1) Cast fc1_w (f32 [256,256], row o = N, col w = K) to f16
__global__ void k_cast_w16(const float* __restrict__ w, _Float16* __restrict__ w16, int n) {
    int i = blockIdx.x * blockDim.x + threadIdx.x;
    if (i < n) w16[i] = (_Float16)w[i];
}

// 2) Gaussian Fourier features: xe16[m, w] = sin(p), xe16[m, 128+w] = cos(p)
//    m = b*H + h in [0, 8192), p = x[m,w] * Wf[h,w] * 2pi
__global__ void k_xe(const float* __restrict__ x, const float* __restrict__ Wf,
                     _Float16* __restrict__ xe16) {
    int i = blockIdx.x * blockDim.x + threadIdx.x;          // B*H*W = 1,048,576
    if (i >= Bn * Hn * Wn) return;
    int w = i % Wn;
    int m = i / Wn;
    int h = m % Hn;
    float p = x[(size_t)m * Wn + w] * Wf[h * Wn + w] * TWO_PI_F;
    float s, c;
    __sincosf(p, &s, &c);
    xe16[(size_t)m * En + w]      = (_Float16)s;
    xe16[(size_t)m * En + Wn + w] = (_Float16)c;
}

// 3) fc1 GEMM via WMMA: C[m,n] = sum_k A[m,k]*Wt[n,k] + bias[n]
//    A = xe16 [8192,256] f16 row-major; Wt = w16 [256,256] ( [N,K] row-major )
__global__ __launch_bounds__(128) void k_fc1_wmma(const _Float16* __restrict__ A,
                                                  const _Float16* __restrict__ Wt,
                                                  const float* __restrict__ bias,
                                                  float* __restrict__ C,
                                                  int Mtot, int Ntot, int Ktot) {
    const int lane = threadIdx.x & 31;
    const int wave = threadIdx.x >> 5;
    const int m0 = blockIdx.x * 16;
    const int n0 = blockIdx.y * 64 + wave * 16;
    const int rc  = lane & 15;           // A row / B col within tile
    const int hiA = (lane >> 4) * 16;    // A k base
    const int hiB = (lane >> 4) * 16;    // B k base

    const _Float16* aRow = A  + (size_t)(m0 + rc) * Ktot;
    const _Float16* bRow = Wt + (size_t)(n0 + rc) * Ktot;

    v8f acc = {};
    for (int kk = 0; kk < Ktot; kk += 32) {
        AV a;
        a.v = *(const v16h*)(aRow + kk + hiA);
        v16h bf = *(const v16h*)(bRow + kk + hiB);
        acc = __builtin_amdgcn_wmma_f32_16x16x32_f16(
                  false, a.v, false, bf, (short)0, acc, false, false);
    }
    CV cv; cv.v = acc;
    const int n = n0 + (lane & 15);
    const int mbase = m0 + ((lane >> 4) * 8);
    const float bv = bias[n];
#pragma unroll
    for (int r = 0; r < 8; ++r)
        C[(size_t)(mbase + r) * Ntot + n] = cv.f[r] + bv;
}

// 4) Generic 3x3 'SAME' conv, NCHW
__global__ void k_conv3x3(const float* __restrict__ in, const float* __restrict__ k,
                          const float* __restrict__ bias, float* __restrict__ out,
                          int Bb, int Cin, int Cout, int H, int W) {
    size_t idx = (size_t)blockIdx.x * blockDim.x + threadIdx.x;
    size_t total = (size_t)Bb * Cout * H * W;
    if (idx >= total) return;
    int x  = (int)(idx % W);
    int y  = (int)((idx / W) % H);
    int co = (int)((idx / ((size_t)W * H)) % Cout);
    int b  = (int)(idx / ((size_t)W * H * Cout));
    float acc = bias[co];
    for (int ci = 0; ci < Cin; ++ci) {
        const float* src = in + (((size_t)b * Cin + ci) * H) * W;
        const float* kc  = k + ((size_t)co * Cin + ci) * 9;
#pragma unroll
        for (int dy = -1; dy <= 1; ++dy) {
            int yy = y + dy;
            if (yy < 0 || yy >= H) continue;
#pragma unroll
            for (int dx = -1; dx <= 1; ++dx) {
                int xx = x + dx;
                if (xx < 0 || xx >= W) continue;
                acc += src[(size_t)yy * W + xx] * kc[(dy + 1) * 3 + (dx + 1)];
            }
        }
    }
    out[idx] = acc;
}

// 5) GroupNorm stats: one block per (b, g). Group channels are contiguous.
//    mean/rstd written at [b*ostride + ooff + g].
__global__ __launch_bounds__(256) void k_gn_stats(const float* __restrict__ in,
                                                  int C, int HW, int G,
                                                  float* __restrict__ mean,
                                                  float* __restrict__ rstd,
                                                  int ostride, int ooff) {
    int b = blockIdx.x / G, g = blockIdx.x % G;
    int cpg = C / G;
    size_t n = (size_t)cpg * HW;
    const float* base = in + ((size_t)b * C + (size_t)g * cpg) * HW;
    float s = 0.f, s2 = 0.f;
    for (size_t i = threadIdx.x; i < n; i += blockDim.x) {
        float v = base[i];
        s += v; s2 += v * v;
    }
    __shared__ float sh[256], sh2[256];
    sh[threadIdx.x] = s; sh2[threadIdx.x] = s2;
    __syncthreads();
    for (int off = 128; off > 0; off >>= 1) {
        if ((int)threadIdx.x < off) {
            sh[threadIdx.x]  += sh[threadIdx.x + off];
            sh2[threadIdx.x] += sh2[threadIdx.x + off];
        }
        __syncthreads();
    }
    if (threadIdx.x == 0) {
        float m   = sh[0] / (float)n;
        float var = sh2[0] / (float)n - m * m;
        int o = b * ostride + ooff + g;
        mean[o] = m;
        rstd[o] = rsqrtf(var + GN_EPS);
    }
}

// 6) Fused GN-normalize + ReLU + 2x2 maxpool. out [B,C,H/2,W/2]
__global__ void k_gn_relu_pool(const float* __restrict__ in,
                               const float* __restrict__ mean, const float* __restrict__ rstd,
                               const float* __restrict__ gamma, const float* __restrict__ beta,
                               float* __restrict__ out, int Bb, int C, int H, int W, int G) {
    size_t idx = (size_t)blockIdx.x * blockDim.x + threadIdx.x;
    int Ho = H >> 1, Wo = W >> 1;
    size_t total = (size_t)Bb * C * Ho * Wo;
    if (idx >= total) return;
    int pw = (int)(idx % Wo);
    int ph = (int)((idx / Wo) % Ho);
    int c  = (int)((idx / ((size_t)Wo * Ho)) % C);
    int b  = (int)(idx / ((size_t)Wo * Ho * C));
    int g  = c / (C / G);
    float m = mean[b * G + g], r = rstd[b * G + g], ga = gamma[c], be = beta[c];
    const float* src = in + (((size_t)b * C + c) * H) * W;
    float best = 0.f;   // post-ReLU values are >= 0
#pragma unroll
    for (int dy = 0; dy < 2; ++dy)
#pragma unroll
        for (int dx = 0; dx < 2; ++dx) {
            float v = src[(size_t)(2 * ph + dy) * W + (2 * pw + dx)];
            v = fmaxf((v - m) * r * ga + be, 0.f);
            best = fmaxf(best, v);
        }
    out[idx] = best;
}

// 7) SiLU + transpose-conv 2x2/stride2 (24->24). k layout [Cin][Cout][2][2]
__global__ void k_t3(const float* __restrict__ p2, const float* __restrict__ k,
                     const float* __restrict__ bias, float* __restrict__ u3,
                     int Bb, int C, int O, int H, int W) {
    size_t idx = (size_t)blockIdx.x * blockDim.x + threadIdx.x;
    size_t total = (size_t)Bb * O * H * W;
    if (idx >= total) return;
    int x = (int)(idx % W);
    int y = (int)((idx / W) % H);
    int o = (int)((idx / ((size_t)W * H)) % O);
    int b = (int)(idx / ((size_t)W * H * O));
    float a00 = bias[o], a01 = a00, a10 = a00, a11 = a00;
    for (int c = 0; c < C; ++c) {
        float v = p2[(((size_t)b * C + c) * H + y) * W + x];
        v = v / (1.f + __expf(-v));                        // SiLU
        const float* kc = k + ((size_t)c * O + o) * 4;
        a00 += v * kc[0]; a01 += v * kc[1];
        a10 += v * kc[2]; a11 += v * kc[3];
    }
    float* dst = u3 + (((size_t)b * O + o) * (2 * H) + 2 * y) * (2 * W) + 2 * x;
    dst[0] = a00; dst[1] = a01;
    dst[2 * W] = a10; dst[2 * W + 1] = a11;
}

// 8) Fused GN3-normalize(concat[c2,u3]) + tconv 2x2 (48->12)
__global__ void k_t4(const float* __restrict__ c2, const float* __restrict__ u3,
                     const float* __restrict__ mean, const float* __restrict__ rstd,
                     const float* __restrict__ gamma, const float* __restrict__ beta,
                     const float* __restrict__ k, const float* __restrict__ bias,
                     float* __restrict__ u4, int Bb, int O, int H, int W) {
    size_t idx = (size_t)blockIdx.x * blockDim.x + threadIdx.x;
    size_t total = (size_t)Bb * O * H * W;
    if (idx >= total) return;
    int x = (int)(idx % W);
    int y = (int)((idx / W) % H);
    int o = (int)((idx / ((size_t)W * H)) % O);
    int b = (int)(idx / ((size_t)W * H * O));
    float a00 = bias[o], a01 = a00, a10 = a00, a11 = a00;
    for (int c = 0; c < 48; ++c) {
        const float* src = (c < 24) ? c2 : u3;
        int cc = (c < 24) ? c : c - 24;
        float raw = src[(((size_t)b * 24 + cc) * H + y) * W + x];
        int g = c >> 3;
        float v = (raw - mean[b * 6 + g]) * rstd[b * 6 + g] * gamma[c] + beta[c];
        const float* kc = k + ((size_t)c * O + o) * 4;
        a00 += v * kc[0]; a01 += v * kc[1];
        a10 += v * kc[2]; a11 += v * kc[3];
    }
    float* dst = u4 + (((size_t)b * O + o) * (2 * H) + 2 * y) * (2 * W) + 2 * x;
    dst[0] = a00; dst[1] = a01;
    dst[2 * W] = a10; dst[2 * W + 1] = a11;
}

// 9) Fused GN4-normalize(concat[c1,u4]) + 1x1 conv (18->6)
__global__ void k_t5(const float* __restrict__ c1, const float* __restrict__ u4,
                     const float* __restrict__ mean, const float* __restrict__ rstd,
                     const float* __restrict__ gamma, const float* __restrict__ beta,
                     const float* __restrict__ k5, const float* __restrict__ bias,
                     float* __restrict__ u5, int Bb, int H, int W) {
    size_t idx = (size_t)blockIdx.x * blockDim.x + threadIdx.x;
    size_t total = (size_t)Bb * 6 * H * W;
    if (idx >= total) return;
    int x = (int)(idx % W);
    int y = (int)((idx / W) % H);
    int o = (int)((idx / ((size_t)W * H)) % 6);
    int b = (int)(idx / ((size_t)W * H * 6));
    float acc = bias[o];
    for (int c = 0; c < 18; ++c) {
        const float* src = (c < 6) ? c1 : u4;
        int cc = (c < 6) ? c : c - 6;
        int Csrc = (c < 6) ? 6 : 12;
        float raw = src[(((size_t)b * Csrc + cc) * H + y) * W + x];
        int g = c / 3;
        float v = (raw - mean[b * 6 + g]) * rstd[b * 6 + g] * gamma[c] + beta[c];
        acc += v * k5[o * 18 + c];
    }
    u5[idx] = acc;
}

// 10) Fused GN5 + final 1x1 conv (6->1) + divide by marginal std(t)
__global__ void k_final(const float* __restrict__ u5,
                        const float* __restrict__ mean, const float* __restrict__ rstd,
                        const float* __restrict__ gamma, const float* __restrict__ beta,
                        const float* __restrict__ fin_k, const float* __restrict__ fin_b,
                        const float* __restrict__ t, float* __restrict__ out,
                        int Bb, int H, int W) {
    size_t idx = (size_t)blockIdx.x * blockDim.x + threadIdx.x;
    size_t total = (size_t)Bb * H * W;
    if (idx >= total) return;
    int b = (int)(idx / ((size_t)W * H));
    float acc = fin_b[0];
#pragma unroll
    for (int c = 0; c < 6; ++c) {
        float raw = u5[((size_t)b * 6 + c) * H * W + (idx % ((size_t)H * W))];
        float v = (raw - mean[b * 6 + c]) * rstd[b * 6 + c] * gamma[c] + beta[c];
        acc += v * fin_k[c];
    }
    float tt = t[b];
    const float ln_sigma = 3.2188758248682006f;           // ln(25)
    float stdv = sqrtf((__expf(2.f * tt * ln_sigma) - 1.f) / (2.f * ln_sigma));
    out[idx] = acc / stdv;
}

// ---------------------------------------------------------------------------
extern "C" void kernel_launch(void* const* d_in, const int* in_sizes, int n_in,
                              void* d_out, int out_size, void* d_ws, size_t ws_size,
                              hipStream_t stream) {
    (void)in_sizes; (void)n_in; (void)out_size; (void)ws_size;

    const float* x      = (const float*)d_in[0];
    const float* t      = (const float*)d_in[1];
    const float* Wf     = (const float*)d_in[2];
    const float* fc1_w  = (const float*)d_in[3];
    const float* fc1_b  = (const float*)d_in[4];
    const float* c1k    = (const float*)d_in[5];
    const float* c1b    = (const float*)d_in[6];
    const float* gn1g   = (const float*)d_in[7];
    const float* gn1b   = (const float*)d_in[8];
    const float* c2k    = (const float*)d_in[9];
    const float* c2b    = (const float*)d_in[10];
    const float* gn2g   = (const float*)d_in[11];
    const float* gn2b   = (const float*)d_in[12];
    const float* t3k    = (const float*)d_in[13];
    const float* t3b    = (const float*)d_in[14];
    const float* gn3g   = (const float*)d_in[15];
    const float* gn3b   = (const float*)d_in[16];
    const float* t4k    = (const float*)d_in[17];
    const float* t4b    = (const float*)d_in[18];
    const float* gn4g   = (const float*)d_in[19];
    const float* gn4b   = (const float*)d_in[20];
    const float* t5k    = (const float*)d_in[21];
    const float* t5b    = (const float*)d_in[22];
    const float* gn5g   = (const float*)d_in[23];
    const float* gn5b   = (const float*)d_in[24];
    const float* fink   = (const float*)d_in[25];
    const float* finb   = (const float*)d_in[26];
    float* out = (float*)d_out;

    // ---- workspace layout (bytes) ----
    char* ws = (char*)d_ws;
    const size_t M = (size_t)Bn * Hn;             // 8192
    size_t off = 0;
    auto take = [&](size_t bytes) { size_t o = off; off += (bytes + 255) & ~(size_t)255; return o; };
    _Float16* xe16 = (_Float16*)(ws + take(M * En * 2));                  // 4 MB
    _Float16* w16  = (_Float16*)(ws + take((size_t)En * En * 2));         // 128 KB
    float* xf = (float*)(ws + take(M * En * 4));                          // 8 MB   [B,1,128,256]
    float* c1 = (float*)(ws + take((size_t)Bn * 6  * 128 * 256 * 4));     // 50 MB
    float* p1 = (float*)(ws + take((size_t)Bn * 6  * 64  * 128 * 4));     // 12.6 MB
    float* c2 = (float*)(ws + take((size_t)Bn * 24 * 64  * 128 * 4));     // 50 MB
    float* p2 = (float*)(ws + take((size_t)Bn * 24 * 32  * 64  * 4));     // 12.6 MB
    size_t off_u3 = take((size_t)Bn * 24 * 64 * 128 * 4);                 // 50 MB
    float* u3 = (float*)(ws + off_u3);
    float* u4 = (float*)(ws + take((size_t)Bn * 12 * 128 * 256 * 4));     // 100 MB
    float* u5 = (float*)(ws + off_u3);                                    // reuse (u3 dead after t4)
    float* stats = (float*)(ws + take(8192 * 4));
    float *mean1 = stats,        *rstd1 = stats + 384;
    float *mean2 = stats + 768,  *rstd2 = stats + 2304;
    float *mean3 = stats + 3840, *rstd3 = stats + 4224;
    float *mean4 = stats + 4608, *rstd4 = stats + 4992;
    float *mean5 = stats + 5376, *rstd5 = stats + 5760;

    const int TB = 256;

    // 1-2) embed prep
    k_cast_w16<<<cdiv(En * En, TB), TB, 0, stream>>>(fc1_w, w16, En * En);
    k_xe<<<cdiv((long long)Bn * Hn * Wn, TB), TB, 0, stream>>>(x, Wf, xe16);

    // 3) fc1 GEMM via WMMA (M=8192, N=256, K=256)
    k_fc1_wmma<<<dim3((int)(M / 16), En / 64), 128, 0, stream>>>(xe16, w16, fc1_b, xf,
                                                                 (int)M, En, En);

    // conv1 (1->6) on [B,1,128,256]
    k_conv3x3<<<cdiv((long long)Bn * 6 * 128 * 256, TB), TB, 0, stream>>>(
        xf, c1k, c1b, c1, Bn, 1, 6, 128, 256);
    // gn1 stats + fused norm/relu/pool -> p1 [B,6,64,128]
    k_gn_stats<<<Bn * 6, 256, 0, stream>>>(c1, 6, 128 * 256, 6, mean1, rstd1, 6, 0);
    k_gn_relu_pool<<<cdiv((long long)Bn * 6 * 64 * 128, TB), TB, 0, stream>>>(
        c1, mean1, rstd1, gn1g, gn1b, p1, Bn, 6, 128, 256, 6);

    // conv2 (6->24) on [B,6,64,128]
    k_conv3x3<<<cdiv((long long)Bn * 24 * 64 * 128, TB), TB, 0, stream>>>(
        p1, c2k, c2b, c2, Bn, 6, 24, 64, 128);
    // gn2 stats + fused norm/relu/pool -> p2 [B,24,32,64]
    k_gn_stats<<<Bn * 24, 256, 0, stream>>>(c2, 24, 64 * 128, 24, mean2, rstd2, 24, 0);
    k_gn_relu_pool<<<cdiv((long long)Bn * 24 * 32 * 64, TB), TB, 0, stream>>>(
        c2, mean2, rstd2, gn2g, gn2b, p2, Bn, 24, 64, 128, 24);

    // SiLU + tconv3 (24->24) -> u3 [B,24,64,128]
    k_t3<<<cdiv((long long)Bn * 24 * 32 * 64, TB), TB, 0, stream>>>(
        p2, t3k, t3b, u3, Bn, 24, 24, 32, 64);

    // gn3 stats over concat[c2,u3]: 6 groups of 8ch -> 3 from c2, 3 from u3
    k_gn_stats<<<Bn * 3, 256, 0, stream>>>(c2, 24, 64 * 128, 3, mean3, rstd3, 6, 0);
    k_gn_stats<<<Bn * 3, 256, 0, stream>>>(u3, 24, 64 * 128, 3, mean3, rstd3, 6, 3);

    // fused gn3-normalize + tconv4 (48->12) -> u4 [B,12,128,256]
    k_t4<<<cdiv((long long)Bn * 12 * 64 * 128, TB), TB, 0, stream>>>(
        c2, u3, mean3, rstd3, gn3g, gn3b, t4k, t4b, u4, Bn, 12, 64, 128);

    // gn4 stats over concat[c1,u4]: 6 groups of 3ch -> 2 from c1, 4 from u4
    k_gn_stats<<<Bn * 2, 256, 0, stream>>>(c1, 6, 128 * 256, 2, mean4, rstd4, 6, 0);
    k_gn_stats<<<Bn * 4, 256, 0, stream>>>(u4, 12, 128 * 256, 4, mean4, rstd4, 6, 2);

    // fused gn4-normalize + 1x1 conv (18->6) -> u5 [B,6,128,256]
    k_t5<<<cdiv((long long)Bn * 6 * 128 * 256, TB), TB, 0, stream>>>(
        c1, u4, mean4, rstd4, gn4g, gn4b, t5k, t5b, u5, Bn, 128, 256);

    // gn5 stats + fused final conv + /std(t)
    k_gn_stats<<<Bn * 6, 256, 0, stream>>>(u5, 6, 128 * 256, 6, mean5, rstd5, 6, 0);
    k_final<<<cdiv((long long)Bn * 128 * 256, TB), TB, 0, stream>>>(
        u5, mean5, rstd5, gn5g, gn5b, fink, finb, t, out, Bn, 128, 256);
}
